// DNC_44538810859771
// MI455X (gfx1250) — matrix operand order repaired and checked by reading, not compiled, not run on hardware
//
#include <hip/hip_runtime.h>
#include <hip/hip_bf16.h>
#include <math.h>

#define B_   32
#define L_   128
#define E_   512
#define H_   1024
#define M_   1024
#define HS_  256
#define R_   4
#define O_   4096
#define CI_  1536
#define KG_  2560      // CI + H (fused LSTM input)
#define NG_  4096      // 4*H
#define NH_  1793      // rk(1024)+wk(256)+er(256)+ad(256)+ws(1)
#define NHP_ 1808      // padded to 16*113
#define KP_  2048      // H + R*HS
#define EPS_ 1e-5f

// heads row offsets
#define RK_OFF_ 0
#define WK_OFF_ 1024
#define ER_OFF_ 1280
#define AD_OFF_ 1536
#define WS_OFF_ 1792

typedef __bf16 bf16_t;
typedef __attribute__((ext_vector_type(8)))  bf16_t v8bf;
typedef __attribute__((ext_vector_type(16))) bf16_t v16bf;
typedef __attribute__((ext_vector_type(8)))  float  v8f;

// ---------------- helpers ----------------

__device__ __forceinline__ unsigned short f2bf(float f) {
    unsigned int u = __float_as_uint(f);
    unsigned int r = u + 0x7FFFu + ((u >> 16) & 1u);   // round-to-nearest-even
    return (unsigned short)(r >> 16);
}

__device__ __forceinline__ float sigmoidf_(float x) {
    return 1.0f / (1.0f + __expf(-x));
}

__device__ __forceinline__ float wave_sum(float v) {
    #pragma unroll
    for (int off = 16; off > 0; off >>= 1) v += __shfl_xor(v, off, 32);
    return v;
}

// block reductions for blockDim.x == 256
__device__ __forceinline__ float block_sum256(float v, float* sh) {
    int tid = threadIdx.x;
    sh[tid] = v; __syncthreads();
    #pragma unroll
    for (int s = 128; s > 0; s >>= 1) { if (tid < s) sh[tid] += sh[tid + s]; __syncthreads(); }
    float r = sh[0]; __syncthreads();
    return r;
}
__device__ __forceinline__ float block_max256(float v, float* sh) {
    int tid = threadIdx.x;
    sh[tid] = v; __syncthreads();
    #pragma unroll
    for (int s = 128; s > 0; s >>= 1) { if (tid < s) sh[tid] = fmaxf(sh[tid], sh[tid + s]); __syncthreads(); }
    float r = sh[0]; __syncthreads();
    return r;
}

// ---------------- setup kernels ----------------

__global__ void k_zero(float* p, size_t n) {
    for (size_t i = (size_t)blockIdx.x * blockDim.x + threadIdx.x; i < n;
         i += (size_t)gridDim.x * blockDim.x) p[i] = 0.0f;
}

// fused LSTM weight: row n = [W_ih[n,0:1536] | W_hh[n,0:1024]]  -> bf16 (NG_ x KG_)
__global__ void k_conv_wg(const float* __restrict__ wih, const float* __restrict__ whh,
                          unsigned short* __restrict__ wg) {
    size_t total = (size_t)NG_ * KG_;
    for (size_t i = (size_t)blockIdx.x * 256 + threadIdx.x; i < total; i += (size_t)gridDim.x * 256) {
        size_t n = i / KG_, k = i % KG_;
        float v = (k < CI_) ? wih[n * CI_ + k] : whh[n * H_ + (k - CI_)];
        wg[i] = f2bf(v);
    }
}

// fused heads weight (NHP_ x H_) bf16, rows >= NH_ zero
__global__ void k_conv_wh(const float* __restrict__ wrk, const float* __restrict__ wwk,
                          const float* __restrict__ wer, const float* __restrict__ wad,
                          const float* __restrict__ wws, unsigned short* __restrict__ wh) {
    size_t total = (size_t)NHP_ * H_;
    for (size_t i = (size_t)blockIdx.x * 256 + threadIdx.x; i < total; i += (size_t)gridDim.x * 256) {
        size_t n = i / H_, k = i % H_;
        float v;
        if      (n < 1024) v = wrk[n * H_ + k];
        else if (n < 1536) v = wwk[(n - 1024) * H_ + k];
        else if (n < 1792) v = (n < ER_OFF_ + 256 && n >= ER_OFF_) ? wer[(n - ER_OFF_) * H_ + k]
                                                                   : wad[(n - AD_OFF_) * H_ + k];
        else if (n == WS_OFF_) v = wws[k];
        else v = 0.0f;
        // fix ranges explicitly (er: 1280..1535, ad: 1536..1791)
        if (n >= 1280 && n < 1536) v = wer[(n - 1280) * H_ + k];
        else if (n >= 1536 && n < 1792) v = wad[(n - 1536) * H_ + k];
        wh[i] = f2bf(v);
    }
}

__global__ void k_conv_wp(const float* __restrict__ wp_f, unsigned short* __restrict__ wp) {
    size_t total = (size_t)O_ * KP_;
    for (size_t i = (size_t)blockIdx.x * 256 + threadIdx.x; i < total; i += (size_t)gridDim.x * 256)
        wp[i] = f2bf(wp_f[i]);
}

__global__ void k_bias_g(const float* __restrict__ bih, const float* __restrict__ bhh,
                         float* __restrict__ bg) {
    int n = blockIdx.x * 256 + threadIdx.x;
    if (n < NG_) bg[n] = bih[n] + bhh[n];
}

__global__ void k_bias_h(const float* __restrict__ brk, const float* __restrict__ bwk,
                         const float* __restrict__ ber, const float* __restrict__ bad,
                         const float* __restrict__ bws, float* __restrict__ bh) {
    int n = blockIdx.x * 256 + threadIdx.x;
    if (n >= NHP_) return;
    float v;
    if      (n < 1024)     v = brk[n];
    else if (n < 1280)     v = bwk[n - 1024];
    else if (n < 1536)     v = ber[n - 1280];
    else if (n < 1792)     v = bad[n - 1536];
    else if (n == WS_OFF_) v = bws[0];
    else v = 0.0f;
    bh[n] = v;
}

// ---------------- WMMA GEMM: C(32 x N) = A(32 x K) * W(N x K)^T + bias ----------------
// A, W in bf16 row-major; one wave per 16x16 C tile; 4 waves per block on n-tiles.
__global__ void k_gemm_bf16(const unsigned short* __restrict__ Aus,
                            const unsigned short* __restrict__ Wus,
                            const float* __restrict__ bias,
                            float* __restrict__ C,
                            int Ntiles, int K, long long crow_stride) {
    int wv   = threadIdx.x >> 5;
    int lane = threadIdx.x & 31;
    int ntile = blockIdx.x * 4 + wv;
    if (ntile >= Ntiles) return;
    int n0 = ntile << 4;
    int m0 = blockIdx.y << 4;
    int l15 = lane & 15;
    bool hi = lane >= 16;

    const bf16_t* A = reinterpret_cast<const bf16_t*>(Aus);
    const bf16_t* W = reinterpret_cast<const bf16_t*>(Wus);
    const bf16_t* arow = A + (size_t)(m0 + l15) * K;   // A fragment: row = m0 + lane%16
    const bf16_t* wrow = W + (size_t)(n0 + l15) * K;   // B fragment: col = n0 + lane%16

    const int aoff = hi ? 16 : 0;
    const int boff = hi ? 16 : 0;
    v8f acc = {};
    for (int k = 0; k < K; k += 32) {
        v8bf a0 = *reinterpret_cast<const v8bf*>(arow + k + aoff);
        v8bf a1 = *reinterpret_cast<const v8bf*>(arow + k + aoff + 8);
        v16bf av = __builtin_shufflevector(a0, a1, 0, 1, 2, 3, 4, 5, 6, 7,
                                                   8, 9, 10, 11, 12, 13, 14, 15);
        v16bf bv = *reinterpret_cast<const v16bf*>(wrow + k + boff);
        acc = __builtin_amdgcn_wmma_f32_16x16x32_bf16(false, av, false, bv,
                                                      (short)0, acc, false, false);
    }
    float bz = bias ? bias[n0 + l15] : 0.0f;
    int mbase = m0 + (hi ? 8 : 0);
    #pragma unroll
    for (int r = 0; r < 8; ++r)
        C[(size_t)(mbase + r) * (size_t)crow_stride + (size_t)(n0 + l15)] = acc[r] + bz;
}

// ---------------- per-step kernels ----------------

// ci = LN([x_t, rv]) -> bf16 ; append raw h as bf16  (row = KG_ = 2560)
__global__ void k_prep_ci(const float* __restrict__ x_emb, int t,
                          const float* __restrict__ rv, const float* __restrict__ hst,
                          const float* __restrict__ g, const float* __restrict__ bb,
                          unsigned short* __restrict__ cih) {
    __shared__ float red[256];
    int b = blockIdx.x, tid = threadIdx.x;
    float v[6]; float s = 0.f, s2 = 0.f;
    #pragma unroll
    for (int q = 0; q < 6; ++q) {
        int j = tid + q * 256;
        float x = (j < E_) ? x_emb[((size_t)b * L_ + t) * E_ + j] : rv[b * (R_ * HS_) + (j - E_)];
        v[q] = x; s += x; s2 += x * x;
    }
    s  = block_sum256(s,  red);
    s2 = block_sum256(s2, red);
    float mean = s * (1.0f / CI_);
    float rstd = rsqrtf(s2 * (1.0f / CI_) - mean * mean + EPS_);
    #pragma unroll
    for (int q = 0; q < 6; ++q) {
        int j = tid + q * 256;
        cih[(size_t)b * KG_ + j] = f2bf((v[q] - mean) * rstd * g[j] + bb[j]);
    }
    #pragma unroll
    for (int q = 0; q < 4; ++q) {
        int j = tid + q * 256;
        cih[(size_t)b * KG_ + CI_ + j] = f2bf(hst[b * H_ + j]);
    }
}

// LSTM cell + hn = LN(h2)
__global__ void k_lstm_ln(const float* __restrict__ gates,
                          float* __restrict__ hst, float* __restrict__ cst,
                          const float* __restrict__ g, const float* __restrict__ bb,
                          float* __restrict__ hnf, unsigned short* __restrict__ hnbf) {
    __shared__ float red[256];
    int b = blockIdx.x, tid = threadIdx.x;
    const float* grow = gates + (size_t)b * NG_;
    float h2[4]; float s = 0.f, s2 = 0.f;
    #pragma unroll
    for (int q = 0; q < 4; ++q) {
        int n = tid + q * 256;
        float gi = grow[n], gf = grow[n + H_], gg = grow[n + 2 * H_], go = grow[n + 3 * H_];
        float c  = cst[b * H_ + n];
        float c2 = sigmoidf_(gf) * c + sigmoidf_(gi) * tanhf(gg);
        cst[b * H_ + n] = c2;
        float hh = sigmoidf_(go) * tanhf(c2);
        hst[b * H_ + n] = hh;
        h2[q] = hh; s += hh; s2 += hh * hh;
    }
    s  = block_sum256(s,  red);
    s2 = block_sum256(s2, red);
    float mean = s * (1.0f / H_);
    float rstd = rsqrtf(s2 * (1.0f / H_) - mean * mean + EPS_);
    #pragma unroll
    for (int q = 0; q < 4; ++q) {
        int n = tid + q * 256;
        float nv = (h2[q] - mean) * rstd * g[n] + bb[n];
        hnf[b * H_ + n]  = nv;
        hnbf[b * H_ + n] = f2bf(nv);
    }
}

// LN of read/write keys + sigmoid/tanh heads
__global__ void k_heads_post(const float* __restrict__ heads,
                             const float* __restrict__ rkg, const float* __restrict__ rkb,
                             const float* __restrict__ wkg, const float* __restrict__ wkb,
                             float* __restrict__ rkn, float* __restrict__ wkn,
                             float* __restrict__ er, float* __restrict__ ad,
                             float* __restrict__ wstr) {
    __shared__ float red[256];
    int b = blockIdx.x, y = blockIdx.y, tid = threadIdx.x;
    const float* hrow = heads + (size_t)b * NHP_;
    if (y < 5) {
        int off = (y < 4) ? (RK_OFF_ + y * HS_) : WK_OFF_;
        float v = hrow[off + tid];
        float s  = block_sum256(v,     red);
        float s2 = block_sum256(v * v, red);
        float mean = s * (1.0f / HS_);
        float rstd = rsqrtf(s2 * (1.0f / HS_) - mean * mean + EPS_);
        const float* g  = (y < 4) ? rkg : wkg;
        const float* bb = (y < 4) ? rkb : wkb;
        float nv = (v - mean) * rstd * g[tid] + bb[tid];
        if (y < 4) rkn[(size_t)(b * R_ + y) * HS_ + tid] = nv;
        else       wkn[(size_t)b * HS_ + tid] = nv;
    } else {
        er[b * HS_ + tid] = sigmoidf_(hrow[ER_OFF_ + tid]);
        ad[b * HS_ + tid] = tanhf(hrow[AD_OFF_ + tid]);
        if (tid == 0) wstr[b] = sigmoidf_(hrow[WS_OFF_]);
    }
}

// write scores: s[b,m] = dot(wkn[b], LN(mem[b,m]));  one wave per slot m
__global__ void k_write_scores(const float* __restrict__ mem, const float* __restrict__ wkn,
                               const float* __restrict__ g, const float* __restrict__ bb,
                               float* __restrict__ swr) {
    int b = blockIdx.y;
    int m = blockIdx.x * 8 + (threadIdx.x >> 5);
    int lane = threadIdx.x & 31;
    const float* row = mem + ((size_t)(b * M_ + m)) * HS_ + lane * 8;
    float x[8]; float s = 0.f, s2 = 0.f;
    #pragma unroll
    for (int i = 0; i < 8; ++i) { x[i] = row[i]; s += x[i]; s2 += x[i] * x[i]; }
    s  = wave_sum(s);
    s2 = wave_sum(s2);
    float mean = s * (1.0f / HS_);
    float rstd = rsqrtf(s2 * (1.0f / HS_) - mean * mean + EPS_);
    float d = 0.f;
    #pragma unroll
    for (int i = 0; i < 8; ++i) {
        int h = lane * 8 + i;
        d += ((x[i] - mean) * rstd * g[h] + bb[h]) * wkn[b * HS_ + h];
    }
    d = wave_sum(d);
    if (lane == 0) swr[(size_t)b * M_ + m] = d;
}

// ww = softmax_m(swr) * wstr
__global__ void k_softmax_write(const float* __restrict__ swr, const float* __restrict__ wstr,
                                float* __restrict__ ww) {
    __shared__ float red[256];
    int b = blockIdx.x, tid = threadIdx.x;
    float v[4]; float mx = -3.4e38f;
    #pragma unroll
    for (int q = 0; q < 4; ++q) { v[q] = swr[(size_t)b * M_ + tid + q * 256]; mx = fmaxf(mx, v[q]); }
    mx = block_max256(mx, red);
    float e[4]; float ps = 0.f;
    #pragma unroll
    for (int q = 0; q < 4; ++q) { e[q] = __expf(v[q] - mx); ps += e[q]; }
    ps = block_sum256(ps, red);
    float sc = wstr[b] / ps;
    #pragma unroll
    for (int q = 0; q < 4; ++q) ww[(size_t)b * M_ + tid + q * 256] = e[q] * sc;
}

// mem = mem*(1 - ww*erase) + ww*add
__global__ void k_mem_update(float* __restrict__ mem, const float* __restrict__ ww,
                             const float* __restrict__ er, const float* __restrict__ ad) {
    size_t total = (size_t)B_ * M_ * HS_;
    for (size_t i = (size_t)blockIdx.x * 256 + threadIdx.x; i < total; i += (size_t)gridDim.x * 256) {
        size_t h = i & (HS_ - 1);
        size_t bm = i >> 8;           // HS_ = 256
        size_t b  = bm >> 10;         // M_  = 1024
        float w = ww[bm];
        float v = mem[i];
        mem[i] = v * (1.0f - w * er[b * HS_ + h]) + w * ad[b * HS_ + h];
    }
}

// read scores: srd[b,r,m] = dot(rkn[b,r], LN(mem2[b,m]))
__global__ void k_read_scores(const float* __restrict__ mem, const float* __restrict__ rkn,
                              const float* __restrict__ g, const float* __restrict__ bb,
                              float* __restrict__ srd) {
    int b = blockIdx.y;
    int m = blockIdx.x * 8 + (threadIdx.x >> 5);
    int lane = threadIdx.x & 31;
    const float* row = mem + ((size_t)(b * M_ + m)) * HS_ + lane * 8;
    float x[8]; float s = 0.f, s2 = 0.f;
    #pragma unroll
    for (int i = 0; i < 8; ++i) { x[i] = row[i]; s += x[i]; s2 += x[i] * x[i]; }
    s  = wave_sum(s);
    s2 = wave_sum(s2);
    float mean = s * (1.0f / HS_);
    float rstd = rsqrtf(s2 * (1.0f / HS_) - mean * mean + EPS_);
    float nv[8];
    #pragma unroll
    for (int i = 0; i < 8; ++i) {
        int h = lane * 8 + i;
        nv[i] = (x[i] - mean) * rstd * g[h] + bb[h];
    }
    #pragma unroll
    for (int r = 0; r < R_; ++r) {
        const float* kv = rkn + ((size_t)(b * R_ + r)) * HS_ + lane * 8;
        float d = 0.f;
        #pragma unroll
        for (int i = 0; i < 8; ++i) d += nv[i] * kv[i];
        d = wave_sum(d);
        if (lane == 0) srd[((size_t)(b * R_ + r)) * M_ + m] = d;
    }
}

// per (b,r): softmax over m then rvec = sum_m p[m] * mem2[b,m,:]
__global__ void k_read_rvec(const float* __restrict__ srd, const float* __restrict__ mem,
                            float* __restrict__ rv) {
    __shared__ float sp[1024];
    __shared__ float red[256];
    int br = blockIdx.x;
    int b = br >> 2, r = br & 3;
    int tid = threadIdx.x;
    const float* s = srd + (size_t)br * M_;
    float v[4]; float mx = -3.4e38f;
    #pragma unroll
    for (int q = 0; q < 4; ++q) { v[q] = s[tid + q * 256]; mx = fmaxf(mx, v[q]); }
    mx = block_max256(mx, red);
    float ps = 0.f;
    #pragma unroll
    for (int q = 0; q < 4; ++q) { float e = __expf(v[q] - mx); sp[tid + q * 256] = e; ps += e; }
    ps = block_sum256(ps, red);     // contains __syncthreads, sp visible after
    float inv = 1.0f / ps;
    int h = tid;
    const float* mb = mem + ((size_t)b * M_) * HS_ + h;
    float acc = 0.f;
    for (int m = 0; m < M_; ++m) acc += sp[m] * mb[(size_t)m * HS_];
    rv[(size_t)b * (R_ * HS_) + r * HS_ + h] = acc * inv;
}

// out = LN([hn, rvec]) -> bf16
__global__ void k_outcat_ln(const float* __restrict__ hnf, const float* __restrict__ rv,
                            const float* __restrict__ g, const float* __restrict__ bb,
                            unsigned short* __restrict__ oc) {
    __shared__ float red[256];
    int b = blockIdx.x, tid = threadIdx.x;
    float v[8]; float s = 0.f, s2 = 0.f;
    #pragma unroll
    for (int q = 0; q < 8; ++q) {
        int j = tid + q * 256;
        float x = (j < H_) ? hnf[b * H_ + j] : rv[b * (R_ * HS_) + (j - H_)];
        v[q] = x; s += x; s2 += x * x;
    }
    s  = block_sum256(s,  red);
    s2 = block_sum256(s2, red);
    float mean = s * (1.0f / KP_);
    float rstd = rsqrtf(s2 * (1.0f / KP_) - mean * mean + EPS_);
    #pragma unroll
    for (int q = 0; q < 8; ++q) {
        int j = tid + q * 256;
        oc[(size_t)b * KP_ + j] = f2bf((v[q] - mean) * rstd * g[j] + bb[j]);
    }
}

// copy mem, h, c to output tail
__global__ void k_final_copy(const float* __restrict__ mem, const float* __restrict__ h,
                             const float* __restrict__ c, float* __restrict__ out_tail) {
    size_t memN = (size_t)B_ * M_ * HS_;
    size_t hN   = (size_t)B_ * H_;
    size_t total = memN + 2 * hN;
    for (size_t i = (size_t)blockIdx.x * 256 + threadIdx.x; i < total; i += (size_t)gridDim.x * 256) {
        float v;
        if (i < memN)            v = mem[i];
        else if (i < memN + hN)  v = h[i - memN];
        else                     v = c[i - memN - hN];
        out_tail[i] = v;
    }
}

// ---------------- host ----------------

extern "C" void kernel_launch(void* const* d_in, const int* in_sizes, int n_in,
                              void* d_out, int out_size, void* d_ws, size_t ws_size,
                              hipStream_t stream) {
    (void)in_sizes; (void)n_in; (void)out_size; (void)ws_size;

    const float* x_emb   = (const float*)d_in[0];
    const float* ln_in_g = (const float*)d_in[1];
    const float* ln_in_b = (const float*)d_in[2];
    const float* W_ih    = (const float*)d_in[3];
    const float* W_hh    = (const float*)d_in[4];
    const float* b_ih    = (const float*)d_in[5];
    const float* b_hh    = (const float*)d_in[6];
    const float* ln_c_g  = (const float*)d_in[7];
    const float* ln_c_b  = (const float*)d_in[8];
    const float* W_rk    = (const float*)d_in[9];
    const float* b_rk    = (const float*)d_in[10];
    const float* W_wk    = (const float*)d_in[11];
    const float* b_wk    = (const float*)d_in[12];
    const float* W_ws    = (const float*)d_in[13];
    const float* b_ws    = (const float*)d_in[14];
    const float* W_er    = (const float*)d_in[15];
    const float* b_er    = (const float*)d_in[16];
    const float* W_ad    = (const float*)d_in[17];
    const float* b_ad    = (const float*)d_in[18];
    const float* ln_rk_g = (const float*)d_in[19];
    const float* ln_rk_b = (const float*)d_in[20];
    const float* ln_wk_g = (const float*)d_in[21];
    const float* ln_wk_b = (const float*)d_in[22];
    const float* ln_m_g  = (const float*)d_in[23];
    const float* ln_m_b  = (const float*)d_in[24];
    const float* ln_po_g = (const float*)d_in[25];
    const float* ln_po_b = (const float*)d_in[26];
    const float* W_proj  = (const float*)d_in[27];
    const float* b_proj  = (const float*)d_in[28];
    float* out = (float*)d_out;

    size_t off = 0;
    auto alloc = [&](size_t bytes) -> void* {
        void* p = (void*)((char*)d_ws + off);
        off += (bytes + 255) & ~(size_t)255;
        return p;
    };

    unsigned short* wg   = (unsigned short*)alloc((size_t)NG_ * KG_ * 2);
    float*          bg   = (float*)         alloc((size_t)NG_ * 4);
    unsigned short* wh   = (unsigned short*)alloc((size_t)NHP_ * H_ * 2);
    float*          bh   = (float*)         alloc((size_t)NHP_ * 4);
    unsigned short* wp   = (unsigned short*)alloc((size_t)O_ * KP_ * 2);
    float*          mem  = (float*)         alloc((size_t)B_ * M_ * HS_ * 4);
    float*          hst  = (float*)         alloc((size_t)B_ * H_ * 4);
    float*          cst  = (float*)         alloc((size_t)B_ * H_ * 4);
    float*          rv   = (float*)         alloc((size_t)B_ * R_ * HS_ * 4);
    unsigned short* cih  = (unsigned short*)alloc((size_t)B_ * KG_ * 2);
    float*          gates= (float*)         alloc((size_t)B_ * NG_ * 4);
    float*          hnf  = (float*)         alloc((size_t)B_ * H_ * 4);
    unsigned short* hnbf = (unsigned short*)alloc((size_t)B_ * H_ * 2);
    float*          heads= (float*)         alloc((size_t)B_ * NHP_ * 4);
    float*          wkn  = (float*)         alloc((size_t)B_ * HS_ * 4);
    float*          rkn  = (float*)         alloc((size_t)B_ * R_ * HS_ * 4);
    float*          ers  = (float*)         alloc((size_t)B_ * HS_ * 4);
    float*          adt  = (float*)         alloc((size_t)B_ * HS_ * 4);
    float*          wstr = (float*)         alloc((size_t)B_ * 4);
    float*          swr  = (float*)         alloc((size_t)B_ * M_ * 4);
    float*          ww   = (float*)         alloc((size_t)B_ * M_ * 4);
    float*          srd  = (float*)         alloc((size_t)B_ * R_ * M_ * 4);
    unsigned short* oc   = (unsigned short*)alloc((size_t)B_ * KP_ * 2);

    // ---- one-time setup (per call; deterministic) ----
    k_conv_wg<<<4096, 256, 0, stream>>>(W_ih, W_hh, wg);
    k_conv_wh<<<2048, 256, 0, stream>>>(W_rk, W_wk, W_er, W_ad, W_ws, wh);
    k_conv_wp<<<4096, 256, 0, stream>>>(W_proj, wp);
    k_bias_g<<<16, 256, 0, stream>>>(b_ih, b_hh, bg);
    k_bias_h<<<8, 256, 0, stream>>>(b_rk, b_wk, b_er, b_ad, b_ws, bh);
    k_zero<<<2048, 256, 0, stream>>>(mem, (size_t)B_ * M_ * HS_);
    k_zero<<<64, 256, 0, stream>>>(hst, (size_t)B_ * H_);
    k_zero<<<64, 256, 0, stream>>>(cst, (size_t)B_ * H_);
    k_zero<<<64, 256, 0, stream>>>(rv,  (size_t)B_ * R_ * HS_);

    const dim3 blkG(128);
    const dim3 grdGates(NG_ / 16 / 4, 2);          // 64 x 2
    const dim3 grdHeads((113 + 3) / 4, 2);         // 29 x 2
    const dim3 grdProj(O_ / 16 / 4, 2);            // 64 x 2
    const dim3 grdScore(M_ / 8, B_);               // 128 x 32

    for (int t = 0; t < L_; ++t) {
        k_prep_ci<<<B_, 256, 0, stream>>>(x_emb, t, rv, hst, ln_in_g, ln_in_b, cih);
        k_gemm_bf16<<<grdGates, blkG, 0, stream>>>(cih, wg, bg, gates, NG_ / 16, KG_, NG_);
        k_lstm_ln<<<B_, 256, 0, stream>>>(gates, hst, cst, ln_c_g, ln_c_b, hnf, hnbf);
        k_gemm_bf16<<<grdHeads, blkG, 0, stream>>>(hnbf, wh, bh, heads, 113, H_, NHP_);
        k_heads_post<<<dim3(B_, 6), 256, 0, stream>>>(heads, ln_rk_g, ln_rk_b,
                                                      ln_wk_g, ln_wk_b,
                                                      rkn, wkn, ers, adt, wstr);
        k_write_scores<<<grdScore, 256, 0, stream>>>(mem, wkn, ln_m_g, ln_m_b, swr);
        k_softmax_write<<<B_, 256, 0, stream>>>(swr, wstr, ww);
        k_mem_update<<<2048, 256, 0, stream>>>(mem, ww, ers, adt);
        k_read_scores<<<grdScore, 256, 0, stream>>>(mem, rkn, ln_m_g, ln_m_b, srd);
        k_read_rvec<<<B_ * R_, 256, 0, stream>>>(srd, mem, rv);
        k_outcat_ln<<<B_, 256, 0, stream>>>(hnf, rv, ln_po_g, ln_po_b, oc);
        k_gemm_bf16<<<grdProj, blkG, 0, stream>>>(oc, wp, b_proj,
                                                  out + (size_t)t * O_,
                                                  O_ / 16, KP_, (long long)L_ * O_);
    }

    k_final_copy<<<4096, 256, 0, stream>>>(mem, hst, cst, out + (size_t)B_ * L_ * O_);
}
